// Net_42588895707765
// MI455X (gfx1250) — compile-verified, matrix operand order checked by reading, not run on hardware
//
#include <hip/hip_runtime.h>
#include <hip/hip_bf16.h>
#include <stdint.h>

typedef __attribute__((ext_vector_type(16))) _Float16 v16h;
typedef __attribute__((ext_vector_type(8)))  float    v8f;

#define H_DIM 8192
#define N_DIM 4096
#define C_DIM 128
#define T_STEPS 256
#define PW 8

// ---------------- d_out layout (float offsets) ----------------
// error[4096], esc[1], r1[8192], r2[4096], r3[128], W1[8192*4096], W2[4096*8192], W3[128*4096]
#define OFF_ERR 0ull
#define OFF_ESC 4096ull
#define OFF_R1  4097ull
#define OFF_R2  12289ull
#define OFF_R3  16385ull
#define OFF_W1  16513ull
#define OFF_W2  33570945ull
#define OFF_W3  67125377ull
// scratch carved out of not-yet-final output regions (aligned up):
#define OFF_Z1   16516ull      // [T][H] f32 (8MB) in W1 region; consumed before W1_new written
#define OFF_XSW  33570952ull   // 1,048,576 f16 swizzled spike matrix in W2 region
#define OFF_IMGT 67125380ull   // in W3 region (w3_update runs last)
#define OFF_HIDM (OFF_IMGT + 4096ull)
#define OFF_ZETA (OFF_IMGT + 12288ull)
#define OFF_R1A  (OFF_IMGT + 20480ull)   // 16B-aligned copy of final r1
#define OFF_R2A  (OFF_IMGT + 28672ull)   // 16B-aligned copy of final r2

__device__ __forceinline__ float stdpf(float dt) {
  float v = 0.01f * __expf(-fabsf(dt) * 0.05f);   // A * exp(-|dt|/20)
  return dt >= 0.f ? v : -v;
}

__device__ __forceinline__ void amax2(float& v, int& i, float ov, int oi) {
  if (ov > v || (ov == v && oi < i)) { v = ov; i = oi; }
}

// ---------------- kernel 1: spike encoding ----------------
// Builds WMMA-B-ready swizzled f16 spike matrix Xsw[kc][tc][lane][e] and img_t.
__global__ __launch_bounds__(256)
void encode_kernel(const float* __restrict__ image, _Float16* __restrict__ Xsw,
                   float* __restrict__ imgt) {
  int idx = blockIdx.x * 256 + threadIdx.x;          // 0 .. 1,048,575
  int e    = idx & 15;
  int lane = (idx >> 4) & 31;
  int tc   = (idx >> 9) & 15;
  int kc   = idx >> 13;
  int k = kc * 32 + ((e < 8) ? e : e + 8) + ((lane & 16) ? 8 : 0);
  int t = tc * 16 + (lane & 15);
  float st = floorf(256.f * image[k]);
  float ft = (float)t;
  Xsw[idx] = (_Float16)((ft >= st && ft < st + (float)PW) ? 1.f : 0.f);
  if (idx < N_DIM) imgt[idx] = 256.f * image[idx];
}

// ---------------- kernel 2 (optional): transpose W2 for coalesced column gather ----
__global__ __launch_bounds__(256)
void w2_transpose_kernel(const float* __restrict__ W2, float* __restrict__ W2t) {
  __shared__ float tile[32][33];
  int x = blockIdx.x * 32 + threadIdx.x;   // h
  int y = blockIdx.y * 32 + threadIdx.y;   // i
#pragma unroll
  for (int j = 0; j < 32; j += 8)
    tile[threadIdx.y + j][threadIdx.x] = W2[(size_t)(y + j) * H_DIM + x];
  __syncthreads();
  int x2 = blockIdx.y * 32 + threadIdx.x;  // i
  int y2 = blockIdx.x * 32 + threadIdx.y;  // h
#pragma unroll
  for (int j = 0; j < 32; j += 8)
    W2t[(size_t)(y2 + j) * N_DIM + x2] = tile[threadIdx.x][threadIdx.y + j];
}

// ---------------- kernel 3: Z1 = W1 @ X via split-f16 WMMA ----------------
// Block = 4 waves; wave w computes rows [bx*64+16w, +16) x cols [by*64, +64) (4 C tiles).
// W1 fp32 tile split into hi/lo f16 so the f32 accumulation is ~exact.
__global__ __launch_bounds__(128)
void gemm_z1_kernel(const float* __restrict__ W1, const _Float16* __restrict__ Xsw,
                    float* __restrict__ Z1t) {
  __shared__ float ldsC[4][64][17];
  const int wave = threadIdx.x >> 5;
  const int lane = threadIdx.x & 31;
  const int hi   = lane >> 4;
  const int m0   = blockIdx.x * 64 + wave * 16;
  const int t0   = blockIdx.y * 64;
  const int row  = m0 + (lane & 15);

  v8f acc0 = {}, acc1 = {}, acc2 = {}, acc3 = {};

  const float* arow = W1 + (size_t)row * N_DIM + hi * 8;
  for (int kc = 0; kc < N_DIM / 32; ++kc) {
    const float* ap = arow + kc * 32;
    __builtin_prefetch(ap + 32, 0, 0);               // global_prefetch_b8 for next K chunk
    float4 a0 = *(const float4*)(ap + 0);
    float4 a1 = *(const float4*)(ap + 4);
    float4 a2 = *(const float4*)(ap + 16);
    float4 a3 = *(const float4*)(ap + 20);
    float av[16] = {a0.x, a0.y, a0.z, a0.w, a1.x, a1.y, a1.z, a1.w,
                    a2.x, a2.y, a2.z, a2.w, a3.x, a3.y, a3.z, a3.w};
    v16h ahi, alo;
#pragma unroll
    for (int e = 0; e < 16; ++e) {
      float w = av[e];
      _Float16 h = (_Float16)w;
      ahi[e] = h;
      alo[e] = (_Float16)(w - (float)h);
    }
    const size_t bbase = ((size_t)(kc * 16 + (t0 >> 4)) * 32 + lane) * 16;
    v16h b0 = *(const v16h*)(Xsw + bbase);
    v16h b1 = *(const v16h*)(Xsw + bbase + 512);
    v16h b2 = *(const v16h*)(Xsw + bbase + 1024);
    v16h b3 = *(const v16h*)(Xsw + bbase + 1536);
    acc0 = __builtin_amdgcn_wmma_f32_16x16x32_f16(false, ahi, false, b0, (short)0, acc0, false, false);
    acc0 = __builtin_amdgcn_wmma_f32_16x16x32_f16(false, alo, false, b0, (short)0, acc0, false, false);
    acc1 = __builtin_amdgcn_wmma_f32_16x16x32_f16(false, ahi, false, b1, (short)0, acc1, false, false);
    acc1 = __builtin_amdgcn_wmma_f32_16x16x32_f16(false, alo, false, b1, (short)0, acc1, false, false);
    acc2 = __builtin_amdgcn_wmma_f32_16x16x32_f16(false, ahi, false, b2, (short)0, acc2, false, false);
    acc2 = __builtin_amdgcn_wmma_f32_16x16x32_f16(false, alo, false, b2, (short)0, acc2, false, false);
    acc3 = __builtin_amdgcn_wmma_f32_16x16x32_f16(false, ahi, false, b3, (short)0, acc3, false, false);
    acc3 = __builtin_amdgcn_wmma_f32_16x16x32_f16(false, alo, false, b3, (short)0, acc3, false, false);
  }

  // C layout: VGPR v, lanes 0-15 -> M=v, N=lane; lanes 16-31 -> M=v+8, N=lane-16.
  const int tl16 = lane & 15;
#pragma unroll
  for (int v = 0; v < 8; ++v) {
    ldsC[wave][0 * 16 + tl16][v + 8 * hi] = acc0[v];
    ldsC[wave][1 * 16 + tl16][v + 8 * hi] = acc1[v];
    ldsC[wave][2 * 16 + tl16][v + 8 * hi] = acc2[v];
    ldsC[wave][3 * 16 + tl16][v + 8 * hi] = acc3[v];
  }
  __syncthreads();
  // Transposed store: Z1t[t][h] so the time-loop reads one contiguous row per step.
#pragma unroll
  for (int r = 0; r < 2; ++r) {
    int tl = lane * 2 + r;
    float* dst = Z1t + (size_t)(t0 + tl) * H_DIM + m0;
    float4 p0, p1, p2, p3;
    p0.x = ldsC[wave][tl][0];  p0.y = ldsC[wave][tl][1];  p0.z = ldsC[wave][tl][2];  p0.w = ldsC[wave][tl][3];
    p1.x = ldsC[wave][tl][4];  p1.y = ldsC[wave][tl][5];  p1.z = ldsC[wave][tl][6];  p1.w = ldsC[wave][tl][7];
    p2.x = ldsC[wave][tl][8];  p2.y = ldsC[wave][tl][9];  p2.z = ldsC[wave][tl][10]; p2.w = ldsC[wave][tl][11];
    p3.x = ldsC[wave][tl][12]; p3.y = ldsC[wave][tl][13]; p3.z = ldsC[wave][tl][14]; p3.w = ldsC[wave][tl][15];
    ((float4*)dst)[0] = p0; ((float4*)dst)[1] = p1;
    ((float4*)dst)[2] = p2; ((float4*)dst)[3] = p3;
  }
}

// ---------------- kernel 4: sequential SNN time loop (one workgroup) ----------------
__global__ __launch_bounds__(1024)
void snn_loop_kernel(const float* __restrict__ Z1t,
                     const float* __restrict__ W2g, int w2_transposed,
                     const float* __restrict__ W3,
                     const float* __restrict__ thr1_in,
                     const float* __restrict__ thr2_in,
                     const float* __restrict__ thr3_in,
                     const float* __restrict__ imgt,
                     float* __restrict__ out_err, float* __restrict__ out_esc,
                     float* __restrict__ out_r1, float* __restrict__ out_r2,
                     float* __restrict__ out_r3,
                     float* __restrict__ hidm, float* __restrict__ zeta,
                     float* __restrict__ r1a, float* __restrict__ r2a) {
  __shared__ float red_v[1024];
  __shared__ int   red_i[1024];
  __shared__ float s_tmp3[8][128];
  __shared__ float s_mem3[128], s_th3[128], s_r3[128];
  __shared__ int   s_active[4096];
  __shared__ int   s_cnt, bc_j, bc_f1, bc_j3, bc_f3;

  const int tid = threadIdx.x;
  const int wv = tid >> 5, ln = tid & 31;

  // per-thread register state: 8 layer-1 neurons, 4 layer-2 neurons
  float mem1[8], th1[8], r1f[8];
  float mem2[4], r2f[4], th2v[4];
  unsigned hid = 0u;
#pragma unroll
  for (int e = 0; e < 8; ++e) { mem1[e] = 0.f; r1f[e] = 0.f; th1[e] = thr1_in[tid * 8 + e]; }
#pragma unroll
  for (int e = 0; e < 4; ++e) { mem2[e] = 0.f; r2f[e] = 0.f; th2v[e] = thr2_in[tid * 4 + e]; }
  if (tid < C_DIM) { s_mem3[tid] = 0.f; s_th3[tid] = thr3_in[tid]; s_r3[tid] = 0.f; }
  __syncthreads();

  for (int t = 0; t < T_STEPS; ++t) {
    // -------- layer 1: leak + drive + local argmax --------
    const float4* z4 = (const float4*)(Z1t + (size_t)t * H_DIM) + tid * 2;
    float4 za = z4[0], zb = z4[1];
    float zv[8] = {za.x, za.y, za.z, za.w, zb.x, zb.y, zb.z, zb.w};
    float lmax = -3.4e38f; int limx = 0;
#pragma unroll
    for (int e = 0; e < 8; ++e) {
      float m = 0.9f * mem1[e] + zv[e];
      mem1[e] = m;
      if (m > lmax) { lmax = m; limx = tid * 8 + e; }
    }
#pragma unroll
    for (int off = 16; off > 0; off >>= 1) {
      float ov = __shfl_xor(lmax, off);
      int   oi = __shfl_xor(limx, off);
      amax2(lmax, limx, ov, oi);
    }
    if (ln == 0) { red_v[wv] = lmax; red_i[wv] = limx; }
    __syncthreads();                                 // B1
    if (tid < 32) {
      float v = red_v[tid]; int i = red_i[tid];
#pragma unroll
      for (int off = 16; off > 0; off >>= 1) {
        float ov = __shfl_xor(v, off);
        int   oi = __shfl_xor(i, off);
        amax2(v, i, ov, oi);
      }
      if (tid == 0) bc_j = i;
    }
    if (tid == 33) s_cnt = 0;
    __syncthreads();                                 // B2
    const int j1 = bc_j;
    if (tid == (j1 >> 3)) {                          // winner-owner thread decides fire
#pragma unroll
      for (int e = 0; e < 8; ++e)
        if ((j1 & 7) == e) {
          int f = mem1[e] > th1[e];
          bc_f1 = f;
          if (f) {
            mem1[e] = 0.f;                           // reset-to-zero (winner only)
            if (r1f[e] == 0.f) r1f[e] = (float)(t + 1);
            hid |= (1u << e);
          }
        }
    }
    __syncthreads();                                 // B3
    const int f1 = bc_f1;
#pragma unroll
    for (int e = 0; e < 8; ++e) {                    // adaptive threshold, all neurons
      float inc = (f1 && (tid * 8 + e) == j1) ? 5.f : 0.f;
      th1[e] = fminf(fmaxf(th1[e] - 0.05f + inc, 150.f), 400.f);
    }
    // -------- layer 2: spk1 is one-hot -> column gather --------
#pragma unroll
    for (int e = 0; e < 4; ++e) {
      int i = tid * 4 + e;
      float add = 0.f;
      if (f1) add = w2_transposed ? W2g[(size_t)j1 * N_DIM + i]
                                  : W2g[(size_t)i * H_DIM + j1];
      float m = 0.8f * mem2[e] + add;
      if (m > th2v[e]) {
        m -= th2v[e];                                // reset-by-subtract
        if (r2f[e] == 0.f) r2f[e] = (float)(t + 1);
        int p = atomicAdd(&s_cnt, 1);
        s_active[p] = i;
      }
      mem2[e] = m;
    }
    __syncthreads();                                 // B4
    // -------- layer 3: sparse 128 x |active| sum --------
    {
      int cnt = s_cnt;
      int c = tid & 127, part = tid >> 7;
      float s = 0.f;
      for (int l = part; l < cnt; l += 8) s += W3[(size_t)c * N_DIM + s_active[l]];
      s_tmp3[part][c] = s;
    }
    __syncthreads();                                 // B5
    if (tid < C_DIM) {
      float dot = 0.f;
#pragma unroll
      for (int p = 0; p < 8; ++p) dot += s_tmp3[p][tid];
      float m3 = 0.9f * s_mem3[tid] + dot;
      s_mem3[tid] = m3;
      red_v[tid] = m3; red_i[tid] = tid;
    }
    __syncthreads();                                 // B6
    if (tid < 32) {
      float v = red_v[tid]; int i = red_i[tid];
      amax2(v, i, red_v[tid + 32], red_i[tid + 32]);
      amax2(v, i, red_v[tid + 64], red_i[tid + 64]);
      amax2(v, i, red_v[tid + 96], red_i[tid + 96]);
#pragma unroll
      for (int off = 16; off > 0; off >>= 1) {
        float ov = __shfl_xor(v, off);
        int   oi = __shfl_xor(i, off);
        amax2(v, i, ov, oi);
      }
      if (tid == 0) { bc_j3 = i; bc_f3 = (v > s_th3[i]) ? 1 : 0; }
    }
    __syncthreads();                                 // B7
    if (tid < C_DIM) {
      int f3 = bc_f3, j3 = bc_j3;
      if (f3) {
        s_mem3[tid] = 0.f;                           // any spike -> zero all mem3
        if (tid == j3 && s_r3[tid] == 0.f) s_r3[tid] = (float)(t + 1);
      }
      float inc = (f3 && tid == j3) ? 5.f : 0.f;
      s_th3[tid] = fminf(fmaxf(s_th3[tid] - 0.05f + inc, 50.f), 200.f);
    }
    __syncthreads();                                 // B8
  }

  // -------- finalize: first-spike times, masks, error --------
#pragma unroll
  for (int e = 0; e < 8; ++e) {
    int h = tid * 8 + e;
    float mask = ((hid >> e) & 1u) ? 1.f : 0.f;
    float rf = (r1f[e] == 0.f) ? 256.f : r1f[e];
    out_r1[h] = rf;
    r1a[h] = rf;
    hidm[h] = mask;
    zeta[h] = (256.f - rf) * mask * (1.f / 256.f);
  }
  float sq = 0.f;
#pragma unroll
  for (int e = 0; e < 4; ++e) {
    int i = tid * 4 + e;
    float rf = (r2f[e] == 0.f) ? 256.f : r2f[e];
    out_r2[i] = rf;
    r2a[i] = rf;
    float err = (imgt[i] - (rf - 4.f)) * (1.f / 256.f);
    out_err[i] = err;
    sq += err * err;
  }
  if (tid < C_DIM) out_r3[tid] = (s_r3[tid] == 0.f) ? 256.f : s_r3[tid];
#pragma unroll
  for (int off = 16; off > 0; off >>= 1) sq += __shfl_xor(sq, off);
  if (ln == 0) red_v[wv] = sq;
  __syncthreads();
  if (tid < 32) {
    float s = red_v[tid];
#pragma unroll
    for (int off = 16; off > 0; off >>= 1) s += __shfl_xor(s, off);
    if (tid == 0) out_esc[0] = s;
  }
}

// ---------------- kernel 5a: W1 STDP ----------------
__global__ __launch_bounds__(256)
void w1_update_kernel(const float* __restrict__ W1, const float* __restrict__ r1,
                      const float* __restrict__ hidm, const float* __restrict__ imgt,
                      float* __restrict__ W1n) {
  int h = blockIdx.x;
  float rh = r1[h], mh = hidm[h];
  const float* src = W1 + (size_t)h * N_DIM;
  float* dst = W1n + (size_t)h * N_DIM;
  for (int i = threadIdx.x; i < N_DIM; i += 256)
    dst[i] = src[i] + stdpf((rh - imgt[i]) * mh);
}

// ---------------- kernel 5b: W2 STDP ----------------
__global__ __launch_bounds__(256)
void w2_update_kernel(const float* __restrict__ W2, const float* __restrict__ r2a,
                      const float* __restrict__ r1a, const float* __restrict__ zeta,
                      const float* __restrict__ imgt, float* __restrict__ W2n) {
  int i = blockIdx.x;
  float r2i = r2a[i];
  float rho = (r2i - 4.f - imgt[i]) * (1.f / 256.f) + 0.15f;
  const float* src = W2 + (size_t)i * H_DIM;
  float* dst = W2n + (size_t)i * H_DIM;
  for (int h = threadIdx.x; h < H_DIM; h += 256) {
    float w = src[h] + stdpf(r2i - r1a[h]) * rho * zeta[h];
    dst[h] = fminf(fmaxf(w, 0.f), 1.f);
  }
}

// ---------------- kernel 5c: W3 STDP ----------------
__global__ __launch_bounds__(256)
void w3_update_kernel(const float* __restrict__ W3, const float* __restrict__ r3,
                      const float* __restrict__ r2a, float* __restrict__ W3n) {
  int c = blockIdx.x;
  float r3c = r3[c];
  const float* src = W3 + (size_t)c * N_DIM;
  float* dst = W3n + (size_t)c * N_DIM;
  for (int i = threadIdx.x; i < N_DIM; i += 256)
    dst[i] = src[i] + stdpf(r3c - r2a[i]);
}

extern "C" void kernel_launch(void* const* d_in, const int* in_sizes, int n_in,
                              void* d_out, int out_size, void* d_ws, size_t ws_size,
                              hipStream_t stream) {
  const float* image = (const float*)d_in[0];
  const float* W1    = (const float*)d_in[1];
  const float* W2    = (const float*)d_in[2];
  const float* W3    = (const float*)d_in[3];
  const float* thr1  = (const float*)d_in[4];
  const float* thr2  = (const float*)d_in[5];
  const float* thr3  = (const float*)d_in[6];
  (void)in_sizes; (void)n_in; (void)out_size;

  float* out = (float*)d_out;
  float*      Z1t  = out + OFF_Z1;
  _Float16*   Xsw  = (_Float16*)(out + OFF_XSW);
  float*      imgt = out + OFF_IMGT;
  float*      hidm = out + OFF_HIDM;
  float*      zeta = out + OFF_ZETA;
  float*      r1a  = out + OFF_R1A;
  float*      r2a  = out + OFF_R2A;

  // Deterministic choice: transpose W2 into ws iff it fits (fixed per harness config).
  const size_t w2t_bytes = (size_t)N_DIM * H_DIM * sizeof(float);
  const int use_w2t = (ws_size >= w2t_bytes) ? 1 : 0;
  float* W2t = (float*)d_ws;

  encode_kernel<<<4096, 256, 0, stream>>>(image, Xsw, imgt);
  if (use_w2t)
    w2_transpose_kernel<<<dim3(H_DIM / 32, N_DIM / 32), dim3(32, 8), 0, stream>>>(W2, W2t);
  gemm_z1_kernel<<<dim3(H_DIM / 64, T_STEPS / 64), 128, 0, stream>>>(W1, Xsw, Z1t);
  snn_loop_kernel<<<1, 1024, 0, stream>>>(
      Z1t, use_w2t ? W2t : W2, use_w2t, W3, thr1, thr2, thr3, imgt,
      out + OFF_ERR, out + OFF_ESC, out + OFF_R1, out + OFF_R2, out + OFF_R3,
      hidm, zeta, r1a, r2a);
  w1_update_kernel<<<H_DIM, 256, 0, stream>>>(W1, out + OFF_R1, hidm, imgt, out + OFF_W1);
  w2_update_kernel<<<N_DIM, 256, 0, stream>>>(W2, r2a, r1a, zeta, imgt, out + OFF_W2);
  w3_update_kernel<<<C_DIM, 256, 0, stream>>>(W3, out + OFF_R3, r2a, out + OFF_W3);
}
